// Graphormer_8727373545997
// MI455X (gfx1250) — compile-verified, matrix operand order checked by reading, not run on hardware
//
#include <hip/hip_runtime.h>

// Problem constants (match reference)
#define NN      2048
#define EE      65536
#define PPAIRS  1000000
#define LLAY    4
#define HHEAD   8
#define DDIM    256
#define DIN_    128
#define DOUT_   64
#define MAXDEG_ 64
#define MPD_    10

typedef __attribute__((ext_vector_type(16))) __bf16 v16bf;
typedef __attribute__((ext_vector_type(8)))  float  v8f;

// ---------------- WMMA fragment helpers (16x16x32 bf16) ----------------
// 16-bit A-matrix 16x32 (ISA 7.12.2): lanes 0-15 hold M=lane, K={0..7,16..23};
// lanes 16-31 hold M=lane-16, K={8..15,24..31}. Two bf16 per VGPR.
__device__ __forceinline__ int k_of_e(int e, int hi) {
  return (e < 8) ? (e + hi * 8) : (e + 8 + hi * 8);
}

// A fragment: row-major A (M x K), tile at (row_base, kb). Two 16B contiguous chunks.
__device__ __forceinline__ v16bf load_a_frag(const __bf16* __restrict__ A, int lda,
                                             int row_base, int kb, int lane) {
  int r = lane & 15, hi = lane >> 4;
  const __bf16* p = A + (size_t)(row_base + r) * lda + kb;
  v16bf a;
#pragma unroll
  for (int e = 0; e < 16; ++e) a[e] = p[k_of_e(e, hi)];
  return a;
}

// B^T fragment: B stored row-major (N x K); row (col_base+c) is a column of math-B.
__device__ __forceinline__ v16bf load_bt_frag(const __bf16* __restrict__ B, int ldb,
                                              int col_base, int kb, int lane) {
  int c = lane & 15, hi = lane >> 4;
  const __bf16* p = B + (size_t)(col_base + c) * ldb + kb;
  v16bf b;
#pragma unroll
  for (int e = 0; e < 16; ++e) b[e] = p[k_of_e(e, hi)];
  return b;
}

#define WMMA_BF16(A_, B_, C_) \
  __builtin_amdgcn_wmma_f32_16x16x32_bf16(false, (A_), false, (B_), (short)0, (C_), false, false)

// ---------------- small utility kernels ----------------
__global__ void zero_f32_kernel(float* p, long long n) {
  long long i = (long long)blockIdx.x * blockDim.x + threadIdx.x;
  long long step = (long long)gridDim.x * blockDim.x;
  for (; i < n; i += step) p[i] = 0.0f;
}

__global__ void zero_i32_kernel(int* p, int n) {
  int i = blockIdx.x * blockDim.x + threadIdx.x;
  if (i < n) p[i] = 0;
}

__global__ void f32_to_bf16_kernel(const float* __restrict__ s, __bf16* __restrict__ d,
                                   long long n) {
  long long i = (long long)blockIdx.x * blockDim.x + threadIdx.x;
  long long step = (long long)gridDim.x * blockDim.x;
  for (; i < n; i += step) d[i] = (__bf16)s[i];
}

// Batched transpose+convert: src = batch x (rows x cols) f32, dst = batch x (cols x rows) bf16
__global__ void transpose_to_bf16_kernel(const float* __restrict__ src, __bf16* __restrict__ dst,
                                         int rows, int cols, long long total) {
  long long i = (long long)blockIdx.x * blockDim.x + threadIdx.x;
  long long step = (long long)gridDim.x * blockDim.x;
  long long mat = (long long)rows * cols;
  for (; i < total; i += step) {
    long long b = i / mat;
    int rem = (int)(i - b * mat);
    int r = rem / cols, c = rem - r * cols;
    dst[b * mat + (size_t)c * rows + r] = (__bf16)src[i];
  }
}

__global__ void degree_kernel(const int* __restrict__ ei, int* deg) {
  int e = blockIdx.x * blockDim.x + threadIdx.x;
  if (e < EE) atomicAdd(&deg[ei[e]], 1);
}

// h = x @ Win + b_in + z[clip(deg)]
__global__ void node_init_kernel(const float* __restrict__ x, const float* __restrict__ Win,
                                 const float* __restrict__ b_in, const float* __restrict__ z,
                                 const int* __restrict__ deg, float* __restrict__ h) {
  int idx = blockIdx.x * blockDim.x + threadIdx.x;
  if (idx >= NN * DDIM) return;
  int n = idx / DDIM, d = idx % DDIM;
  float acc = b_in[d];
  const float* xr = x + (size_t)n * DIN_;
#pragma unroll 4
  for (int k = 0; k < DIN_; ++k) acc += xr[k] * Win[(size_t)k * DDIM + d];
  int dg = deg[n];
  dg = dg < 0 ? 0 : (dg > MAXDEG_ - 1 ? MAXDEG_ - 1 : dg);
  h[idx] = acc + z[(size_t)dg * DDIM + d];
}

__global__ void scatter_bias_kernel(const int* __restrict__ dist, const int* __restrict__ didx,
                                    const float* __restrict__ bsp, float* __restrict__ bmat) {
  int p = blockIdx.x * blockDim.x + threadIdx.x;
  if (p >= PPAIRS) return;
  int d = dist[p];
  d = d < 0 ? 0 : (d > MPD_ - 1 ? MPD_ - 1 : d);
  bmat[(size_t)didx[p] * NN + didx[PPAIRS + p]] = bsp[d];
}

// LayerNorm row kernel: one block (DDIM threads) per row, bf16 out
__global__ void layernorm_bf16_kernel(const float* __restrict__ X, const float* __restrict__ w,
                                      const float* __restrict__ b, __bf16* __restrict__ Y) {
  __shared__ float red[DDIM];
  int row = blockIdx.x, t = threadIdx.x;
  float xv = X[(size_t)row * DDIM + t];
  red[t] = xv;
  __syncthreads();
  for (int o = DDIM / 2; o > 0; o >>= 1) {
    if (t < o) red[t] += red[t + o];
    __syncthreads();
  }
  float mu = red[0] * (1.0f / DDIM);
  __syncthreads();
  float d = xv - mu;
  red[t] = d * d;
  __syncthreads();
  for (int o = DDIM / 2; o > 0; o >>= 1) {
    if (t < o) red[t] += red[t + o];
    __syncthreads();
  }
  float var = red[0] * (1.0f / DDIM);
  float y = d * rsqrtf(var + 1e-5f) * w[t] + b[t];
  Y[(size_t)row * DDIM + t] = (__bf16)y;
}

// ---------------- generic WMMA GEMM (B pre-transposed) ----------------
// C = A(bf16, MxK row-major) x B^T where B is (Nc x K) row-major.
// One wave computes a 32x64 tile (8 WMMAs: 2 A frags x 4 B frags reuse).
// Nc % 64 == 0, M % 32 == 0, K % 32 == 0.
// Batched over blockIdx.z. Output: f32 (outF) and/or bf16 (outB).
// outT: bf16 output stored transposed: outB[col*ldc + row] (for producing V^T).
__global__ void wmma_gemm_kernel(const __bf16* __restrict__ A, int lda, long long sA,
                                 const __bf16* __restrict__ B, int ldb, long long sB,
                                 const float* __restrict__ bias, long long sBias,
                                 const float* __restrict__ resid, int ldr,
                                 float* __restrict__ outF, __bf16* __restrict__ outB,
                                 int ldc, long long sOut, int outT,
                                 int M, int Nc, int K) {
  int zi = blockIdx.z;
  A += (size_t)zi * sA;
  B += (size_t)zi * sB;
  if (bias) bias += (size_t)zi * sBias;
  size_t oOff = (size_t)zi * sOut;

  int groups_n = Nc >> 6;          // 64-wide column groups
  int groups_m = M >> 5;           // 32-tall row groups
  int wave = (int)((blockIdx.x * blockDim.x + threadIdx.x) >> 5);
  int lane = threadIdx.x & 31;
  if (wave >= groups_m * groups_n) return;  // wave-uniform: EXEC stays all-ones
  int rb0 = (wave / groups_n) * 32;
  int cb0 = (wave % groups_n) * 64;

  v8f a00 = {}, a01 = {}, a02 = {}, a03 = {};
  v8f a10 = {}, a11 = {}, a12 = {}, a13 = {};
  for (int kb = 0; kb < K; kb += 32) {
    v16bf fa0 = load_a_frag(A, lda, rb0, kb, lane);
    v16bf fa1 = load_a_frag(A, lda, rb0 + 16, kb, lane);
    v16bf b0 = load_bt_frag(B, ldb, cb0 +  0, kb, lane);
    v16bf b1 = load_bt_frag(B, ldb, cb0 + 16, kb, lane);
    v16bf b2 = load_bt_frag(B, ldb, cb0 + 32, kb, lane);
    v16bf b3 = load_bt_frag(B, ldb, cb0 + 48, kb, lane);
    a00 = WMMA_BF16(fa0, b0, a00);
    a01 = WMMA_BF16(fa0, b1, a01);
    a02 = WMMA_BF16(fa0, b2, a02);
    a03 = WMMA_BF16(fa0, b3, a03);
    a10 = WMMA_BF16(fa1, b0, a10);
    a11 = WMMA_BF16(fa1, b1, a11);
    a12 = WMMA_BF16(fa1, b2, a12);
    a13 = WMMA_BF16(fa1, b3, a13);
  }

  int c = lane & 15, hi = lane >> 4;
  v8f accs[2][4] = {{a00, a01, a02, a03}, {a10, a11, a12, a13}};
#pragma unroll
  for (int t = 0; t < 4; ++t) {
    int col = cb0 + t * 16 + c;
    float bv = bias ? bias[col] : 0.0f;  // hoisted: one load per column tile
#pragma unroll
    for (int mt = 0; mt < 2; ++mt) {
#pragma unroll
      for (int r = 0; r < 8; ++r) {
        int row = rb0 + mt * 16 + r + hi * 8;
        float v = accs[mt][t][r] + bv;
        if (resid) v += resid[(size_t)row * ldr + col];
        if (outF)  outF[oOff + (size_t)row * ldc + col] = v;
        if (outB) {
          if (outT) outB[oOff + (size_t)col * ldc + row] = (__bf16)v;
          else      outB[oOff + (size_t)row * ldc + col] = (__bf16)v;
        }
      }
    }
  }
}

// ---------------- attention scores: S = (Q K^T * scale + bmat) * mneg ----------------
// One wave computes a 32x64 score tile (2 Q frags x 4 K frags -> 8 WMMAs).
__global__ void wmma_scores_kernel(const __bf16* __restrict__ Q, const __bf16* __restrict__ Km,
                                   const float* __restrict__ bmat, const int* __restrict__ gptr,
                                   float* __restrict__ S, float scale) {
  const int groups_n = NN / 64;
  int wave = (int)((blockIdx.x * blockDim.x + threadIdx.x) >> 5);
  int lane = threadIdx.x & 31;
  if (wave >= (NN / 32) * groups_n) return;
  int rb0 = (wave / groups_n) * 32;
  int cb0 = (wave % groups_n) * 64;

  v8f a00 = {}, a01 = {}, a02 = {}, a03 = {};
  v8f a10 = {}, a11 = {}, a12 = {}, a13 = {};
  for (int kb = 0; kb < DDIM; kb += 32) {
    v16bf fa0 = load_a_frag(Q, DDIM, rb0, kb, lane);
    v16bf fa1 = load_a_frag(Q, DDIM, rb0 + 16, kb, lane);
    v16bf b0 = load_bt_frag(Km, DDIM, cb0 +  0, kb, lane);
    v16bf b1 = load_bt_frag(Km, DDIM, cb0 + 16, kb, lane);
    v16bf b2 = load_bt_frag(Km, DDIM, cb0 + 32, kb, lane);
    v16bf b3 = load_bt_frag(Km, DDIM, cb0 + 48, kb, lane);
    a00 = WMMA_BF16(fa0, b0, a00);
    a01 = WMMA_BF16(fa0, b1, a01);
    a02 = WMMA_BF16(fa0, b2, a02);
    a03 = WMMA_BF16(fa0, b3, a03);
    a10 = WMMA_BF16(fa1, b0, a10);
    a11 = WMMA_BF16(fa1, b1, a11);
    a12 = WMMA_BF16(fa1, b2, a12);
    a13 = WMMA_BF16(fa1, b3, a13);
  }

  int c = lane & 15, hi = lane >> 4;
  v8f accs[2][4] = {{a00, a01, a02, a03}, {a10, a11, a12, a13}};
#pragma unroll
  for (int t = 0; t < 4; ++t) {
    int col = cb0 + t * 16 + c;
    int gc = gptr[col];
#pragma unroll
    for (int mt = 0; mt < 2; ++mt) {
#pragma unroll
      for (int r = 0; r < 8; ++r) {
        int row = rb0 + mt * 16 + r + hi * 8;
        float v = accs[mt][t][r] * scale + bmat[(size_t)row * NN + col];
        float m = (gptr[row] == gc) ? 1.0f : -1.0e6f;
        S[(size_t)row * NN + col] = v * m;
      }
    }
  }
}

// ---------------- row softmax with zero mask, bf16 probs ----------------
__global__ void softmax_mask_kernel(const float* __restrict__ S, const int* __restrict__ gptr,
                                    __bf16* __restrict__ Pr) {
  __shared__ float red[256];
  int row = blockIdx.x, t = threadIdx.x;
  const float* s = S + (size_t)row * NN;
  float v[NN / 256];
  float mx = -3.0e38f;
#pragma unroll
  for (int i = 0; i < NN / 256; ++i) {
    v[i] = s[t + i * 256];
    mx = fmaxf(mx, v[i]);
  }
  red[t] = mx;
  __syncthreads();
  for (int o = 128; o > 0; o >>= 1) {
    if (t < o) red[t] = fmaxf(red[t], red[t + o]);
    __syncthreads();
  }
  mx = red[0];
  __syncthreads();
  float sum = 0.0f;
#pragma unroll
  for (int i = 0; i < NN / 256; ++i) {
    v[i] = __expf(v[i] - mx);
    sum += v[i];
  }
  red[t] = sum;
  __syncthreads();
  for (int o = 128; o > 0; o >>= 1) {
    if (t < o) red[t] += red[t + o];
    __syncthreads();
  }
  float inv = 1.0f / red[0];
  int gr = gptr[row];
  __bf16* pr = Pr + (size_t)row * NN;
#pragma unroll
  for (int i = 0; i < NN / 256; ++i) {
    int col = t + i * 256;
    float p = v[i] * inv;
    if (gptr[col] != gr) p = 0.0f;
    pr[col] = (__bf16)p;
  }
}

// ---------------- host side ----------------
static void run_gemm(hipStream_t stream,
                     const __bf16* A, int lda, long long sA,
                     const __bf16* B, int ldb, long long sB,
                     const float* bias, long long sBias,
                     const float* resid, int ldr,
                     float* outF, __bf16* outB, int ldc, long long sOut, int outT,
                     int M, int Nc, int K, int nz) {
  int waves = (M / 32) * (Nc / 64);
  dim3 grid((waves + 7) / 8, 1, nz);
  wmma_gemm_kernel<<<grid, dim3(256), 0, stream>>>(A, lda, sA, B, ldb, sB,
                                                   bias, sBias, resid, ldr,
                                                   outF, outB, ldc, sOut, outT, M, Nc, K);
}

extern "C" void kernel_launch(void* const* d_in, const int* in_sizes, int n_in,
                              void* d_out, int out_size, void* d_ws, size_t ws_size,
                              hipStream_t stream) {
  const float* x    = (const float*)d_in[0];
  const int*   ei   = (const int*)d_in[1];
  const int*   gptr = (const int*)d_in[2];
  const int*   dist = (const int*)d_in[3];
  const int*   didx = (const int*)d_in[4];
  const float* Win  = (const float*)d_in[5];
  const float* b_in = (const float*)d_in[6];
  const float* ztab = (const float*)d_in[7];
  const float* bsp  = (const float*)d_in[8];
  const float* Wq   = (const float*)d_in[9];
  const float* bq   = (const float*)d_in[10];
  const float* Wk   = (const float*)d_in[11];
  const float* bk   = (const float*)d_in[12];
  const float* Wv   = (const float*)d_in[13];
  const float* bv   = (const float*)d_in[14];
  const float* Wo   = (const float*)d_in[15];
  const float* bo   = (const float*)d_in[16];
  const float* ln1w = (const float*)d_in[17];
  const float* ln1b = (const float*)d_in[18];
  const float* ln2w = (const float*)d_in[19];
  const float* ln2b = (const float*)d_in[20];
  const float* Wff  = (const float*)d_in[21];
  const float* bff  = (const float*)d_in[22];
  const float* Wout = (const float*)d_in[23];
  const float* bout = (const float*)d_in[24];
  float* out = (float*)d_out;
  (void)in_sizes; (void)n_in; (void)out_size; (void)ws_size;

  // ---- workspace carving ----
  char* w = (char*)d_ws;
  auto carve = [&](size_t bytes) {
    void* p = (void*)w;
    w += (bytes + 255) & ~(size_t)255;
    return p;
  };
  int*    deg   = (int*)carve((size_t)NN * 4);
  float*  h     = (float*)carve((size_t)NN * DDIM * 4);
  float*  xp    = (float*)carve((size_t)NN * DDIM * 4);
  __bf16* hn    = (__bf16*)carve((size_t)NN * DDIM * 2);
  __bf16* qb    = (__bf16*)carve((size_t)HHEAD * NN * DDIM * 2);
  __bf16* kb    = (__bf16*)carve((size_t)HHEAD * NN * DDIM * 2);
  __bf16* vT    = (__bf16*)carve((size_t)HHEAD * DDIM * NN * 2);   // V^T per head: D x N
  float*  bmat  = (float*)carve((size_t)NN * NN * 4);
  float*  S     = (float*)carve((size_t)NN * NN * 4);
  __bf16* Pr    = (__bf16*)carve((size_t)NN * NN * 2);
  __bf16* oc    = (__bf16*)carve((size_t)NN * HHEAD * DDIM * 2);
  __bf16* hb    = (__bf16*)carve((size_t)NN * DDIM * 2);
  __bf16* wqT   = (__bf16*)carve((size_t)LLAY * HHEAD * DDIM * DDIM * 2);
  __bf16* wkT   = (__bf16*)carve((size_t)LLAY * HHEAD * DDIM * DDIM * 2);
  __bf16* wvT   = (__bf16*)carve((size_t)LLAY * HHEAD * DDIM * DDIM * 2);
  __bf16* woT   = (__bf16*)carve((size_t)LLAY * HHEAD * DDIM * DDIM * 2);
  __bf16* wffT  = (__bf16*)carve((size_t)LLAY * DDIM * DDIM * 2);
  __bf16* woutT = (__bf16*)carve((size_t)DDIM * DOUT_ * 2);

  // ---- preprocessing ----
  zero_i32_kernel<<<(NN + 255) / 256, 256, 0, stream>>>(deg, NN);
  degree_kernel<<<(EE + 255) / 256, 256, 0, stream>>>(ei, deg);
  node_init_kernel<<<(NN * DDIM + 255) / 256, 256, 0, stream>>>(x, Win, b_in, ztab, deg, h);
  zero_f32_kernel<<<2048, 256, 0, stream>>>(bmat, (long long)NN * NN);
  scatter_bias_kernel<<<(PPAIRS + 255) / 256, 256, 0, stream>>>(dist, didx, bsp, bmat);

  // Transpose+convert all weights to bf16 (N^T x K layout for the B^T operand)
  long long wqn = (long long)LLAY * HHEAD * DDIM * DDIM;
  transpose_to_bf16_kernel<<<2048, 256, 0, stream>>>(Wq, wqT, DDIM, DDIM, wqn);
  transpose_to_bf16_kernel<<<2048, 256, 0, stream>>>(Wk, wkT, DDIM, DDIM, wqn);
  transpose_to_bf16_kernel<<<2048, 256, 0, stream>>>(Wv, wvT, DDIM, DDIM, wqn);
  transpose_to_bf16_kernel<<<2048, 256, 0, stream>>>(Wo, woT, HHEAD * DDIM, DDIM, wqn);
  transpose_to_bf16_kernel<<<512, 256, 0, stream>>>(Wff, wffT, DDIM, DDIM,
                                                    (long long)LLAY * DDIM * DDIM);
  transpose_to_bf16_kernel<<<64, 256, 0, stream>>>(Wout, woutT, DDIM, DOUT_,
                                                   (long long)DDIM * DOUT_);

  const float scale = 0.0625f;  // 1/sqrt(256)
  const long long hdd = (long long)HHEAD * DDIM * DDIM;

  for (int l = 0; l < LLAY; ++l) {
    // hn = LN1(h) -> bf16
    layernorm_bf16_kernel<<<NN, DDIM, 0, stream>>>(h, ln1w + l * DDIM, ln1b + l * DDIM, hn);

    // QKV, batched over heads via grid.z (V stored transposed per head)
    run_gemm(stream, hn, DDIM, 0, wqT + l * hdd, DDIM, (long long)DDIM * DDIM,
             bq + (size_t)l * HHEAD * DDIM, DDIM, nullptr, 0,
             nullptr, qb, DDIM, (long long)NN * DDIM, 0, NN, DDIM, DDIM, HHEAD);
    run_gemm(stream, hn, DDIM, 0, wkT + l * hdd, DDIM, (long long)DDIM * DDIM,
             bk + (size_t)l * HHEAD * DDIM, DDIM, nullptr, 0,
             nullptr, kb, DDIM, (long long)NN * DDIM, 0, NN, DDIM, DDIM, HHEAD);
    run_gemm(stream, hn, DDIM, 0, wvT + l * hdd, DDIM, (long long)DDIM * DDIM,
             bv + (size_t)l * HHEAD * DDIM, DDIM, nullptr, 0,
             nullptr, vT, NN, (long long)DDIM * NN, 1, NN, DDIM, DDIM, HHEAD);

    // attention per head (S / Pr buffers reused)
    for (int hh = 0; hh < HHEAD; ++hh) {
      int score_waves = (NN / 32) * (NN / 64);
      wmma_scores_kernel<<<(score_waves + 7) / 8, 256, 0, stream>>>(
          qb + (size_t)hh * NN * DDIM, kb + (size_t)hh * NN * DDIM, bmat, gptr, S, scale);
      softmax_mask_kernel<<<NN, 256, 0, stream>>>(S, gptr, Pr);
      // O_h = Pr @ V_h  (B operand = V^T, contiguous)  -> bf16 into concat buffer
      run_gemm(stream, Pr, NN, 0, vT + (size_t)hh * DDIM * NN, NN, 0,
               nullptr, 0, nullptr, 0,
               nullptr, oc + (size_t)hh * DDIM, HHEAD * DDIM, 0, 0, NN, DDIM, NN, 1);
    }

    // xp = oc @ Wo[l] + bo[l] + h
    run_gemm(stream, oc, HHEAD * DDIM, 0, woT + l * hdd, HHEAD * DDIM, 0,
             bo + (size_t)l * DDIM, 0, h, DDIM,
             xp, nullptr, DDIM, 0, 0, NN, DDIM, HHEAD * DDIM, 1);

    // h = LN2(xp) @ Wff[l] + bff[l] + xp
    layernorm_bf16_kernel<<<NN, DDIM, 0, stream>>>(xp, ln2w + l * DDIM, ln2b + l * DDIM, hn);
    run_gemm(stream, hn, DDIM, 0, wffT + (size_t)l * DDIM * DDIM, DDIM, 0,
             bff + (size_t)l * DDIM, 0, xp, DDIM,
             h, nullptr, DDIM, 0, 0, NN, DDIM, DDIM, 1);
  }

  // out = h @ Wout + b_out
  f32_to_bf16_kernel<<<512, 256, 0, stream>>>(h, hb, (long long)NN * DDIM);
  run_gemm(stream, hb, DDIM, 0, woutT, DDIM, 0,
           bout, 0, nullptr, 0,
           out, nullptr, DOUT_, 0, 0, NN, DOUT_, DDIM, 1);
}
